// FeatureAlignmentModule_46720654246255
// MI455X (gfx1250) — matrix operand order repaired and checked by reading, not compiled, not run on hardware
//
#include <hip/hip_runtime.h>
#include <math.h>

typedef __attribute__((ext_vector_type(16))) __bf16 v16bf;
typedef __attribute__((ext_vector_type(8)))  float  v8f;

constexpr int B_ = 4, C_ = 128, H_ = 128, W_ = 128;
constexpr int HW_ = H_ * W_;
constexpr int NCH_OFF = 72;   // 9 taps * 8 channel-blocks (Ci = 256)
constexpr int NCH_DEF = 36;   // 9 taps * 4 channel-blocks (Ci = 128)
constexpr size_t PACK_OFF_ELEMS = (size_t)2 * NCH_OFF * 512;  // 2 M-tiles (27 -> 32)
constexpr size_t PACK_DEF_ELEMS = (size_t)8 * NCH_DEF * 512;  // 8 M-tiles (Co = 128)
constexpr size_t WS_OFF_DEF = PACK_OFF_ELEMS * sizeof(unsigned short);
constexpr size_t WS_OFF_O   = WS_OFF_DEF + PACK_DEF_ELEMS * sizeof(unsigned short);

__device__ __forceinline__ unsigned short f2bf_bits(float f) {
    unsigned int u = __float_as_uint(f);
    unsigned int lsb = (u >> 16) & 1u;
    u += 0x7fffu + lsb;                 // round to nearest even
    return (unsigned short)(u >> 16);
}
__device__ __forceinline__ unsigned int pack2bf(float lo, float hi) {
    return (unsigned int)f2bf_bits(lo) | ((unsigned int)f2bf_bits(hi) << 16);
}
__device__ __forceinline__ int clampi(int v, int lo, int hi) {
    return v < lo ? lo : (v > hi ? hi : v);
}

// ---------------------------------------------------------------------------
// Prep: pack fp32 weights into bf16 in a WMMA 16-bit A-matrix
// per-lane layout (contiguous k): element j (of 16 per lane) holds K = j + 16*hi,
// hi = lane/16, M = mtile*16 + lane%16.  One contiguous 32B load/lane later.
// ---------------------------------------------------------------------------
__global__ void prep_pack_weights(const float* __restrict__ offset_w,
                                  const float* __restrict__ deform_w,
                                  unsigned short* __restrict__ packA_off,
                                  unsigned short* __restrict__ packA_def) {
    size_t idx = (size_t)blockIdx.x * 256 + threadIdx.x;
    size_t total = PACK_OFF_ELEMS + PACK_DEF_ELEMS;
    if (idx >= total) return;
    if (idx < PACK_OFF_ELEMS) {
        int mtile = (int)(idx / (NCH_OFF * 512));
        int rem   = (int)(idx % (NCH_OFF * 512));
        int chunk = rem / 512, p = rem % 512;
        int lane = p / 16, j = p % 16, hi = lane / 16;
        int kl = j + 16 * hi;                            // K within 32-chunk
        int tap = chunk / 8, cblk = chunk % 8;
        int c = cblk * 32 + kl;                          // 0..255 (concat channel)
        int m = mtile * 16 + (lane % 16);
        int ty = tap / 3, tx = tap % 3;
        float w = 0.f;
        if (m < 27) w = offset_w[((m * 256 + c) * 3 + ty) * 3 + tx];
        packA_off[idx] = f2bf_bits(w);
    } else {
        size_t id2 = idx - PACK_OFF_ELEMS;
        int mtile = (int)(id2 / (NCH_DEF * 512));
        int rem   = (int)(id2 % (NCH_DEF * 512));
        int chunk = rem / 512, p = rem % 512;
        int lane = p / 16, j = p % 16, hi = lane / 16;
        int kl = j + 16 * hi;
        int tap = chunk / 4, cblk = chunk % 4;
        int c = cblk * 32 + kl;                          // 0..127
        int m = mtile * 16 + (lane % 16);
        int ty = tap / 3, tx = tap % 3;
        float w = deform_w[((m * 128 + c) * 3 + ty) * 3 + tx];
        packA_def[id2] = f2bf_bits(w);
    }
}

// ---------------------------------------------------------------------------
// Kernel A: offset conv (Ci=256 -> 27ch), implicit GEMM via WMMA bf16.
// Block = (b, y); 8 waves, each owns 16 pixels of the row and both M-tiles.
// LDS B tile stored as packed u32 (two bf16/channel-pair) so staging is one
// b32 store per element-pair and the WMMA B fragment is one 32B row load.
// ---------------------------------------------------------------------------
__global__ __launch_bounds__(256)
void offset_conv_kernel(const float* __restrict__ rgb, const float* __restrict__ ir,
                        const unsigned short* __restrict__ packA,
                        const float* __restrict__ offset_b, float* __restrict__ obuf) {
    const int y = blockIdx.x;
    const int b = blockIdx.y;
    const int tid = threadIdx.x;
    const int wid = tid >> 5;
    const int lane = tid & 31;
    const int xbase = wid * 16;
    const int cpair = lane & 15;        // channel pair (k0 = 2*cpair, k0+1)
    const int nh    = lane >> 4;        // pixel half (pixels 8*nh .. 8*nh+7)
    const int rowbase = (cpair >> 3) << 4;   // 16*(k0/16)

    // Row r of stageU[wid] = 8 dwords = 16 bf16 = lane r's WMMA B fragment.
    __shared__ __align__(32) unsigned int stageU[8][32][8];

    v8f acc0 = {0.f, 0.f, 0.f, 0.f, 0.f, 0.f, 0.f, 0.f};
    v8f acc1 = {0.f, 0.f, 0.f, 0.f, 0.f, 0.f, 0.f, 0.f};

    for (int tap = 0; tap < 9; ++tap) {
        const int ty = tap / 3, tx = tap % 3;
        const int yy = y + ty - 1;
        const bool yok = (yy >= 0) && (yy < H_);
        const int yc = clampi(yy, 0, H_ - 1);
        const int xs = xbase + tx - 1;
        for (int cblk = 0; cblk < 8; ++cblk) {
            const int chunk = tap * 8 + cblk;
            const int cin = cblk * 32 + 2 * cpair;            // even concat channel
            const float* base = (cin < 128) ? rgb : ir;       // wave-uniform per cblk
            const float* s0 = base + (size_t)b * C_ * HW_
                                   + (size_t)(cin & 127) * HW_ + (size_t)yc * W_;
            const float* s1 = s0 + HW_;
#pragma unroll
            for (int i = 0; i < 8; ++i) {
                int n = 8 * nh + i;
                int xx = xs + n;
                bool v = yok && ((unsigned)xx < (unsigned)W_);
                int xc = clampi(xx, 0, W_ - 1);
                float f0 = s0[xc];                // clamped, unconditional load
                float f1 = s1[xc];
                f0 = v ? f0 : 0.f;
                f1 = v ? f1 : 0.f;
                stageU[wid][rowbase + n][cpair & 7] = pack2bf(f0, f1);
            }
            __builtin_amdgcn_wave_barrier();      // LDS in-order within wave
            v16bf bfrag;
            __builtin_memcpy(&bfrag, &stageU[wid][lane][0], sizeof(bfrag));
            __builtin_amdgcn_wave_barrier();
            const unsigned short* pA0 = packA + ((size_t)(0 * NCH_OFF + chunk) * 32 + lane) * 16;
            const unsigned short* pA1 = packA + ((size_t)(1 * NCH_OFF + chunk) * 32 + lane) * 16;
            if (chunk + 1 < NCH_OFF)
                __builtin_prefetch(packA + ((size_t)(chunk + 1) * 32 + lane) * 16, 0, 3);
            v16bf a0, a1;
            __builtin_memcpy(&a0, pA0, sizeof(a0));
            __builtin_memcpy(&a1, pA1, sizeof(a1));
            acc0 = __builtin_amdgcn_wmma_f32_16x16x32_bf16(false, a0, false, bfrag,
                                                           (short)0, acc0, false, false);
            acc1 = __builtin_amdgcn_wmma_f32_16x16x32_bf16(false, a1, false, bfrag,
                                                           (short)0, acc1, false, false);
        }
    }

    const int n = lane & 15;
    const int hi8 = (lane >> 4) * 8;
    const int x = xbase + n;
#pragma unroll
    for (int r = 0; r < 8; ++r) {
        int co = r + hi8;                                   // 0..15: raw offset channels
        float v = acc0[r] + offset_b[co];
        obuf[(((size_t)b * 27 + co) * H_ + y) * W_ + x] = v;
        int co1 = 16 + r + hi8;                             // 16..31
        if (co1 < 27) {
            float v1 = acc1[r] + offset_b[co1];
            if (co1 >= 18) v1 = 1.f / (1.f + __expf(-v1));  // pre-sigmoid the mask
            obuf[(((size_t)b * 27 + co1) * H_ + y) * W_ + x] = v1;
        }
    }
}

// ---------------------------------------------------------------------------
// Kernel B: fused deformable sampling + WMMA GEMM (M=128, K=1152).
// Block = (xtile, y, b); 8 waves = 8 M-tiles share ONE sampled B matrix per
// chunk (bilinear work done once, consumed 8x). Per tap, 16 threads
// precompute 4 clamped flat indices + 4 validity*mask-folded bilinear
// weights per pixel, so the per-channel inner loop is 8 loads + 8 FMAs.
// ---------------------------------------------------------------------------
__global__ __launch_bounds__(256)
void deform_conv_kernel(const float* __restrict__ ir,
                        const unsigned short* __restrict__ packA,
                        const float* __restrict__ obuf,
                        const float* __restrict__ deform_b, float* __restrict__ out) {
    const int xtile = blockIdx.x;      // 0..7
    const int y     = blockIdx.y;      // 0..127
    const int b     = blockIdx.z;      // 0..3
    const int tid   = threadIdx.x;
    const int wid   = tid >> 5;        // M-tile
    const int lane  = tid & 31;
    const int xbase = xtile * 16;
    const int cpair = tid & 15;        // channel pair within 32-chunk
    const int np    = tid >> 4;        // pixel 0..15 handled by this thread

    __shared__ __align__(32) unsigned int Bbuf[2][32][8];
    __shared__ int   p_i00[16], p_i01[16], p_i10[16], p_i11[16];
    __shared__ float p_w00[16], p_w01[16], p_w10[16], p_w11[16];

    const float* xb = ir + (size_t)b * C_ * HW_;
    const float* ob = obuf + (size_t)b * 27 * HW_;

    v8f acc = {0.f, 0.f, 0.f, 0.f, 0.f, 0.f, 0.f, 0.f};

    for (int tap = 0; tap < 9; ++tap) {
        if (tid < 16) {                // interpolation params: once per tap
            int n = tid;
            int xs = xbase + n;
            float dy = ob[(size_t)(2 * tap) * HW_ + y * W_ + xs];
            float dx = ob[(size_t)(2 * tap + 1) * HW_ + y * W_ + xs];
            float m  = ob[(size_t)(18 + tap) * HW_ + y * W_ + xs];   // pre-sigmoided
            int ky = tap / 3, kx = tap % 3;
            float py = (float)y + (float)(ky - 1) + dy;
            float px = (float)xs + (float)(kx - 1) + dx;
            float fy = floorf(py), fx = floorf(px);
            int y0 = (int)fy, x0 = (int)fx;
            float wy = py - fy, wx = px - fx;
            bool vy0 = (unsigned)y0 < (unsigned)H_;
            bool vy1 = (unsigned)(y0 + 1) < (unsigned)H_;
            bool vx0 = (unsigned)x0 < (unsigned)W_;
            bool vx1 = (unsigned)(x0 + 1) < (unsigned)W_;
            int yc0 = clampi(y0, 0, H_ - 1), yc1 = clampi(y0 + 1, 0, H_ - 1);
            int xc0 = clampi(x0, 0, W_ - 1), xc1 = clampi(x0 + 1, 0, W_ - 1);
            p_i00[n] = yc0 * W_ + xc0;  p_i01[n] = yc0 * W_ + xc1;
            p_i10[n] = yc1 * W_ + xc0;  p_i11[n] = yc1 * W_ + xc1;
            p_w00[n] = (vy0 && vx0) ? (1.f - wy) * (1.f - wx) * m : 0.f;
            p_w01[n] = (vy0 && vx1) ? (1.f - wy) * wx * m : 0.f;
            p_w10[n] = (vy1 && vx0) ? wy * (1.f - wx) * m : 0.f;
            p_w11[n] = (vy1 && vx1) ? wy * wx * m : 0.f;
        }
        __syncthreads();
        for (int cblk = 0; cblk < 4; ++cblk) {
            const int chunk = tap * 4 + cblk;
            const int cur = chunk & 1;
            // cooperative sampling: this thread -> channel pair (2*cpair, +1)
            // of the 32-chunk at pixel np; one packed b32 LDS store.
            const int c = cblk * 32 + 2 * cpair;
            const float* xc0p = xb + (size_t)c * HW_;
            const float* xc1p = xc0p + HW_;
            int i00 = p_i00[np], i01 = p_i01[np], i10 = p_i10[np], i11 = p_i11[np];
            float w00 = p_w00[np], w01 = p_w01[np], w10 = p_w10[np], w11 = p_w11[np];
            float v0 = xc0p[i00] * w00 + xc0p[i01] * w01 + xc0p[i10] * w10 + xc0p[i11] * w11;
            float v1 = xc1p[i00] * w00 + xc1p[i01] * w01 + xc1p[i10] * w10 + xc1p[i11] * w11;
            Bbuf[cur][((cpair >> 3) << 4) + np][cpair & 7] = pack2bf(v0, v1);
            __syncthreads();
            v16bf bfrag;
            __builtin_memcpy(&bfrag, &Bbuf[cur][lane][0], sizeof(bfrag));
            const unsigned short* pA = packA + ((size_t)(wid * NCH_DEF + chunk) * 32 + lane) * 16;
            if (chunk + 1 < NCH_DEF)
                __builtin_prefetch(packA + ((size_t)(wid * NCH_DEF + chunk + 1) * 32 + lane) * 16, 0, 3);
            v16bf afrag;
            __builtin_memcpy(&afrag, pA, sizeof(afrag));
            acc = __builtin_amdgcn_wmma_f32_16x16x32_bf16(false, afrag, false, bfrag,
                                                          (short)0, acc, false, false);
        }
    }

    const int n = lane & 15;
    const int hi8 = (lane >> 4) * 8;
    const int x = xbase + n;
#pragma unroll
    for (int r = 0; r < 8; ++r) {
        int co = wid * 16 + r + hi8;
        out[(((size_t)b * C_ + co) * H_ + y) * W_ + x] = acc[r] + deform_b[co];
    }
}

extern "C" void kernel_launch(void* const* d_in, const int* in_sizes, int n_in,
                              void* d_out, int out_size, void* d_ws, size_t ws_size,
                              hipStream_t stream) {
    const float* rgb      = (const float*)d_in[0];
    const float* ir       = (const float*)d_in[1];
    const float* offset_w = (const float*)d_in[2];
    const float* offset_b = (const float*)d_in[3];
    const float* deform_w = (const float*)d_in[4];
    const float* deform_b = (const float*)d_in[5];
    float* out = (float*)d_out;

    unsigned short* packA_off = (unsigned short*)d_ws;
    unsigned short* packA_def = (unsigned short*)((char*)d_ws + WS_OFF_DEF);
    float* obuf = (float*)((char*)d_ws + WS_OFF_O);

    size_t total_pack = PACK_OFF_ELEMS + PACK_DEF_ELEMS;
    prep_pack_weights<<<dim3((unsigned)((total_pack + 255) / 256)), 256, 0, stream>>>(
        offset_w, deform_w, packA_off, packA_def);

    dim3 gA(H_, B_);
    offset_conv_kernel<<<gA, 256, 0, stream>>>(rgb, ir, packA_off, offset_b, obuf);

    dim3 gB(W_ / 16, H_, B_);
    deform_conv_kernel<<<gB, 256, 0, stream>>>(ir, packA_def, obuf, deform_b, out);
}
